// Attention_5755256176729
// MI455X (gfx1250) — matrix operand order repaired and checked by reading, not compiled, not run on hardware
//
#include <hip/hip_runtime.h>

// ---------------------------------------------------------------------------
// Shapes (fixed by the reference): B=16, S=1024, E=D=1024
// ---------------------------------------------------------------------------
#define DD 1024
#define BB 16
#define SS 1024

typedef __attribute__((ext_vector_type(16))) __bf16 v16bf;
typedef __attribute__((ext_vector_type(8)))  __bf16 v8bf;
typedef __attribute__((ext_vector_type(8)))  float  v8f;
typedef __attribute__((ext_vector_type(4)))  unsigned int u32x4;
typedef __attribute__((ext_vector_type(4)))  int  i32x4;
typedef __attribute__((ext_vector_type(8)))  int  i32x8;

union V16U { v16bf v; v8bf h[2]; };

// ---------------------------------------------------------------------------
// Tensor Data Mover: 2-D bf16 tile  global -> LDS  (ISA cdna5 §8 D# layout)
// Issue from ONE wave only; completion via s_wait_tensorcnt + block barrier.
// ---------------------------------------------------------------------------
static __device__ __forceinline__ void tdm_load_bf16_2d(
    void* ldsDst, const void* gsrc, unsigned tile0, unsigned tile1,
    unsigned long long stride0 /*elements*/) {
  unsigned long long ga = (unsigned long long)(uintptr_t)gsrc;
  u32x4 g0;
  g0[0] = 1u;                                  // count=1 (valid)
  g0[1] = (unsigned)(uintptr_t)ldsDst;         // lds_addr
  g0[2] = (unsigned)ga;                        // global_addr[31:0]
  g0[3] = (unsigned)((ga >> 32) & 0x1FFFFFFull) | (2u << 30);  // addr[56:32]|type=2
  const unsigned td0 = 0x40000000u;            // huge tensor_dim0: no OOB clamp
  const unsigned td1 = tile1;
  i32x8 g1;
  g1[0] = (int)(1u << 16);                     // workgroup_mask=0, data_size=2B
  g1[1] = (int)((td0 & 0xFFFFu) << 16);
  g1[2] = (int)(((td0 >> 16) & 0xFFFFu) | ((td1 & 0xFFFFu) << 16));
  g1[3] = (int)(((td1 >> 16) & 0xFFFFu) | ((tile0 & 0xFFFFu) << 16));
  g1[4] = (int)(tile1 & 0xFFFFu);
  g1[5] = (int)(unsigned)(stride0 & 0xFFFFFFFFull);
  g1[6] = (int)(unsigned)((stride0 >> 32) & 0xFFFFull);
  g1[7] = 0;
  i32x4 gz = {0, 0, 0, 0};
#if defined(__clang_major__) && (__clang_major__ >= 23)
  i32x8 gz8 = {0, 0, 0, 0, 0, 0, 0, 0};
  __builtin_amdgcn_tensor_load_to_lds(g0, g1, gz, gz, gz8, 0);
#else
  __builtin_amdgcn_tensor_load_to_lds(g0, g1, gz, gz, 0);
#endif
}

// A-operand (16x32 bf16) per-lane fetch from an LDS [16][1024] bf16 panel.
static __device__ __forceinline__ v16bf load_a_lds(const __bf16* rowBase, int k0) {
  V16U u;
  u.h[0] = *(const v8bf*)(rowBase + k0);
  u.h[1] = *(const v8bf*)(rowBase + k0 + 8);
  return u.v;
}

static __device__ __forceinline__ v8f wmma_bf16(v16bf a, v16bf b, v8f c) {
  return __builtin_amdgcn_wmma_f32_16x16x32_bf16(
      false, a, false, b, (short)0, c, false, false);
}

// acc[0..1] += A(16x1024 bf16 LDS) @ W.  Software-pipelined: B loads for k+32
// are issued before the WMMAs for k, so loadcnt waits stay nonzero.
static __device__ __forceinline__ void gemm_pair(v8f acc[2], const __bf16* A,
                                                 const unsigned short* Wu,
                                                 int w, int lane) {
  const __bf16* W = (const __bf16*)Wu;
  const int arow  = lane & 15;
  const int koffA = (lane >> 4) << 4;
  const int koffB = (lane >> 4) << 4;
  const int n0    = (w * 2) * 16 + (lane & 15);
  const __bf16* Wr0 = W + (size_t)n0 * DD + koffB;
  const __bf16* Wr1 = Wr0 + 16 * DD;
  v16bf b0 = *(const v16bf*)(Wr0);
  v16bf b1 = *(const v16bf*)(Wr1);
  for (int kk = 0; kk < DD; kk += 32) {
    const int kn = (kk + 32) & (DD - 1);
    v16bf b0n = *(const v16bf*)(Wr0 + kn);
    v16bf b1n = *(const v16bf*)(Wr1 + kn);
    v16bf a   = load_a_lds(A + arow * DD, kk + koffA);
    acc[0] = wmma_bf16(a, b0, acc[0]);
    acc[1] = wmma_bf16(a, b1, acc[1]);
    b0 = b0n; b1 = b1n;
  }
}

// LayerNorm over the 16x1024 matrix distributed across all waves' acc regs.
// Half-wave shuffle pre-reduction (rows uniform across each 16-lane group),
// then one lane per group does the ds_add_f32 atomics.
static __device__ __forceinline__ void ln_regs(v8f acc[2], int w, int lane,
    const float* __restrict__ lw, const float* __restrict__ lb,
    float* rsum, float* rsq, float* rmean, float* rrstd, __bf16* outBF) {
  const int tid = threadIdx.x;
  if (tid < 16) { rsum[tid] = 0.f; rsq[tid] = 0.f; }
  __syncthreads();
  const int rbase = (lane >> 4) << 3;
#pragma unroll
  for (int i = 0; i < 8; ++i) {
    float a0 = acc[0][i], a1 = acc[1][i];
    float s = a0 + a1;
    float q = a0 * a0 + a1 * a1;
#pragma unroll
    for (int off = 1; off < 16; off <<= 1) {
      s += __shfl_xor(s, off, 16);
      q += __shfl_xor(q, off, 16);
    }
    if ((lane & 15) == 0) {
      atomicAdd(&rsum[rbase + i], s);
      atomicAdd(&rsq[rbase + i], q);
    }
  }
  __syncthreads();
  if (tid < 16) {
    float m   = rsum[tid] * (1.0f / (float)DD);
    float var = rsq[tid] * (1.0f / (float)DD) - m * m;
    rmean[tid] = m;
    rrstd[tid] = rsqrtf(var + 1e-5f);
  }
  __syncthreads();
#pragma unroll
  for (int tt = 0; tt < 2; ++tt) {
    int col = (w * 2 + tt) * 16 + (lane & 15);
    float g = lw[col], bb = lb[col];
#pragma unroll
    for (int i = 0; i < 8; ++i) {
      int row = rbase + i;
      float y = (acc[tt][i] - rmean[row]) * rrstd[row] * g + bb;
      outBF[row * DD + col] = (__bf16)y;
    }
  }
  __syncthreads();
}

// new_state = ln(state @ W1 + out @ W2) @ W3   (state: [16,1024] f32 in global)
static __device__ __forceinline__ void summary_update(
    float* Sg, const unsigned short* W1, const unsigned short* W2,
    const unsigned short* W3, __bf16* outBF, __bf16* sBF,
    const float* lw, const float* lb,
    float* rsum, float* rsq, float* rmean, float* rrstd, int w, int lane) {
  const int tid = threadIdx.x;
  for (int e = tid; e < BB * DD; e += 1024) sBF[e] = (__bf16)Sg[e];
  __syncthreads();
  v8f a2[2] = {};
  gemm_pair(a2, sBF, W1, w, lane);
  gemm_pair(a2, outBF, W2, w, lane);
  ln_regs(a2, w, lane, lw, lb, rsum, rsq, rmean, rrstd, sBF);  // tmp -> sBF bf16
  v8f a3[2] = {};
  gemm_pair(a3, sBF, W3, w, lane);
  const int rbase = (lane >> 4) << 3;
#pragma unroll
  for (int tt = 0; tt < 2; ++tt) {
    int col = (w * 2 + tt) * 16 + (lane & 15);
#pragma unroll
    for (int i = 0; i < 8; ++i) Sg[(rbase + i) * DD + col] = a3[tt][i];
  }
  __syncthreads();
}

// ---------------------------------------------------------------------------
// Prep kernels
// ---------------------------------------------------------------------------
__global__ void cvt_bf16(const float* __restrict__ s, unsigned short* __restrict__ d, int n) {
  int i = blockIdx.x * 256 + threadIdx.x;
  if (i < n) ((__bf16*)d)[i] = (__bf16)s[i];
}

__global__ void cvt_bf16_t(const float* __restrict__ s, unsigned short* __restrict__ d) {
  int n = blockIdx.x * 16 + threadIdx.x;
  int k = blockIdx.y * 16 + threadIdx.y;
  ((__bf16*)d)[(size_t)n * DD + k] = (__bf16)s[(size_t)k * DD + n];
}

__global__ void zero_f32(float* __restrict__ p, int n) {
  int i = blockIdx.x * 256 + threadIdx.x;
  if (i < n) p[i] = 0.0f;
}

// ---------------------------------------------------------------------------
// Bookend GEMM: C[16384,1024] = A_bf16 @ W_bf16 + bias.  A panel TDM-DMAed,
// B stream software-pipelined one k-step ahead.
// ---------------------------------------------------------------------------
template <bool STORE_BF16>
__global__ __launch_bounds__(256) void gemm16k(const unsigned short* __restrict__ Au,
                                               const unsigned short* __restrict__ Wu,
                                               const float* __restrict__ bias,
                                               void* __restrict__ Cout) {
  __shared__ __align__(32) __bf16 aBF[16 * DD];
  const int tid  = threadIdx.x;
  const int m0   = blockIdx.x * 16;
  const int w    = tid >> 5;
  const int lane = tid & 31;
  const int n0   = blockIdx.y * 512 + w * 64;

  if (tid < 32) {  // wave 0: async DMA of the contiguous 32KB A panel
    tdm_load_bf16_2d(aBF, Au + (size_t)m0 * DD, 16 * DD, 1, 16 * DD);
    __builtin_amdgcn_s_wait_tensorcnt(0);
  }
  __syncthreads();

  const __bf16* W = (const __bf16*)Wu;
  const int arow  = lane & 15;
  const int koffA = (lane >> 4) << 4;
  const int koffB = (lane >> 4) << 4;
  const __bf16* Wr[4];
#pragma unroll
  for (int t = 0; t < 4; ++t)
    Wr[t] = W + (size_t)(n0 + t * 16 + (lane & 15)) * DD + koffB;

  v8f acc[4] = {};
  v16bf b[4];
#pragma unroll
  for (int t = 0; t < 4; ++t) b[t] = *(const v16bf*)(Wr[t]);
  for (int kk = 0; kk < DD; kk += 32) {
    const int kn = (kk + 32) & (DD - 1);
    v16bf bn[4];
#pragma unroll
    for (int t = 0; t < 4; ++t) bn[t] = *(const v16bf*)(Wr[t] + kn);
    v16bf a = load_a_lds(aBF + arow * DD, kk + koffA);
#pragma unroll
    for (int t = 0; t < 4; ++t) acc[t] = wmma_bf16(a, b[t], acc[t]);
#pragma unroll
    for (int t = 0; t < 4; ++t) b[t] = bn[t];
  }
  const int rbase = (lane >> 4) << 3;
#pragma unroll
  for (int t = 0; t < 4; ++t) {
    int col  = n0 + t * 16 + (lane & 15);
    float bv = bias[col];
#pragma unroll
    for (int i = 0; i < 8; ++i) {
      int row = rbase + i;
      float v = acc[t][i] + bv;
      if (STORE_BF16)
        ((__bf16*)Cout)[(size_t)(m0 + row) * DD + col] = (__bf16)v;
      else
        ((float*)Cout)[(size_t)(m0 + row) * DD + col] = v;
    }
  }
}

// ---------------------------------------------------------------------------
// Sequential scan: one persistent 1024-thread workgroup (32 waves).
// gs lives as bf16 in LDS; x_t tiles are double-buffered TDM DMAs (the DMA
// for x_{t+1} overlaps all of step t's GEMMs); gss/lss live in global.
// ---------------------------------------------------------------------------
__global__ __launch_bounds__(1024) void scan_kernel(
    const unsigned short* __restrict__ xsu,   // bf16 [B,S,D]
    const unsigned short* Wgsc, const unsigned short* Wgii, const unsigned short* Wgos,
    const unsigned short* Wlsc, const unsigned short* Wlsi, const unsigned short* Wlos,
    const unsigned short* Wssc, const unsigned short* Wssi, const unsigned short* Wsos,
    const float* __restrict__ lnw, const float* __restrict__ lnb,
    float* gss, float* lss,
    unsigned short* __restrict__ outsu) {     // bf16 [B,S,D]
  __shared__ __align__(32) __bf16 gBF[BB * DD];
  __shared__ __align__(32) __bf16 xPing[BB * DD];
  __shared__ __align__(32) __bf16 xPong[BB * DD];
  __shared__ __align__(32) __bf16 sBF[BB * DD];
  __shared__ float rsum[16], rsq[16], rmean[16], rrstd[16];

  const __bf16* xsb = (const __bf16*)xsu;
  __bf16* outsb     = (__bf16*)outsu;
  const int tid   = threadIdx.x;
  const int w     = tid >> 5;
  const int lane  = tid & 31;
  const int rbase = (lane >> 4) << 3;

  for (int e = tid; e < BB * DD; e += 1024) gBF[e] = (__bf16)0.0f;

  // prologue: start DMA of x_0
  if (tid < 32)
    tdm_load_bf16_2d(xPing, xsb, DD, BB, (unsigned long long)SS * DD);
  __syncthreads();

  for (int t = 0; t < SS; ++t) {
    __bf16* xBF = (t & 1) ? xPong : xPing;
    __bf16* xNx = (t & 1) ? xPing : xPong;
    // wait for x_t, then immediately launch the DMA for x_{t+1}
    if (tid < 32) __builtin_amdgcn_s_wait_tensorcnt(0);
    __syncthreads();
    if (tid < 32 && (t + 1) < SS)
      tdm_load_bf16_2d(xNx, xsb + (size_t)(t + 1) * DD, DD, BB,
                       (unsigned long long)SS * DD);

    // gs_pre = gs @ Wgsc + x_t @ Wgii + gss + lss
    v8f acc[2] = {};
    gemm_pair(acc, gBF, Wgsc, w, lane);
    gemm_pair(acc, xBF, Wgii, w, lane);
#pragma unroll
    for (int tt = 0; tt < 2; ++tt) {
      int col = (w * 2 + tt) * 16 + (lane & 15);
#pragma unroll
      for (int i = 0; i < 8; ++i) {
        int row = rbase + i;
        acc[tt][i] += gss[row * DD + col] + lss[row * DD + col];
      }
    }
    // gs = ln(gs_pre) -> bf16 straight into gBF
    ln_regs(acc, w, lane, lnw, lnb, rsum, rsq, rmean, rrstd, gBF);

    // out = gs @ Wgos -> bf16 outs (global) and xBF (A operand for summaries)
    v8f o[2] = {};
    gemm_pair(o, gBF, Wgos, w, lane);
#pragma unroll
    for (int tt = 0; tt < 2; ++tt) {
      int col = (w * 2 + tt) * 16 + (lane & 15);
#pragma unroll
      for (int i = 0; i < 8; ++i) {
        int row = rbase + i;
        float v = o[tt][i];
        __bf16 vb = (__bf16)v;
        outsb[((size_t)row * SS + t) * DD + col] = vb;
        xBF[row * DD + col] = vb;   // x_t consumed; reuse as 'out' panel
      }
    }
    __syncthreads();

    if ((t & 63) == 63)
      summary_update(gss, Wssc, Wssi, Wsos, xBF, sBF, lnw, lnb,
                     rsum, rsq, rmean, rrstd, w, lane);
    if ((t & 15) == 15)
      summary_update(lss, Wlsc, Wlsi, Wlos, xBF, sBF, lnw, lnb,
                     rsum, rsq, rmean, rrstd, w, lane);
  }
}

// ---------------------------------------------------------------------------
// Host launcher
// ---------------------------------------------------------------------------
extern "C" void kernel_launch(void* const* d_in, const int* in_sizes, int n_in,
                              void* d_out, int out_size, void* d_ws, size_t ws_size,
                              hipStream_t stream) {
  (void)in_sizes; (void)n_in; (void)out_size; (void)ws_size;
  const float* emb   = (const float*)d_in[0];
  const float* Wg[9] = {
      (const float*)d_in[1], (const float*)d_in[2], (const float*)d_in[3],   // gsc gii gos
      (const float*)d_in[4], (const float*)d_in[5], (const float*)d_in[6],   // lsc lsi los
      (const float*)d_in[7], (const float*)d_in[8], (const float*)d_in[9]};  // ssc ssi sos
  const float* lnw  = (const float*)d_in[10];
  const float* lnb  = (const float*)d_in[11];
  const float* We2s = (const float*)d_in[12];
  const float* be2s = (const float*)d_in[13];
  const float* Ws2o = (const float*)d_in[14];
  const float* bs2o = (const float*)d_in[15];

  constexpr size_t MB = 1024ull * 1024ull;
  char* ws = (char*)d_ws;
  unsigned short* xsBF   = (unsigned short*)(ws);             // 32 MB bf16 [B,S,D]
  unsigned short* outsBF = (unsigned short*)(ws + 32 * MB);   // 32 MB bf16 [B,S,D]
  unsigned short* embBF  = (unsigned short*)(ws + 64 * MB);   // 32 MB bf16 [B,S,E]
  unsigned short* wE2S   = (unsigned short*)(ws + 96 * MB);   // 2 MB
  unsigned short* wS2O   = (unsigned short*)(ws + 98 * MB);   // 2 MB
  unsigned short* wT[9];
  for (int i = 0; i < 9; ++i)
    wT[i] = (unsigned short*)(ws + 100 * MB + (size_t)i * 2 * MB);
  float* gssb = (float*)(ws + 118 * MB);                      // 64 KB
  float* lssb = gssb + BB * DD;                               // 64 KB

  const int NTOK = BB * SS * DD;  // 16M

  cvt_bf16<<<dim3(NTOK / 256), dim3(256), 0, stream>>>(emb, embBF, NTOK);
  cvt_bf16<<<dim3((DD * DD) / 256), dim3(256), 0, stream>>>(We2s, wE2S, DD * DD);
  cvt_bf16<<<dim3((DD * DD) / 256), dim3(256), 0, stream>>>(Ws2o, wS2O, DD * DD);
  for (int i = 0; i < 9; ++i)
    cvt_bf16_t<<<dim3(DD / 16, DD / 16), dim3(16, 16), 0, stream>>>(Wg[i], wT[i]);
  zero_f32<<<dim3((2 * BB * DD) / 256), dim3(256), 0, stream>>>(gssb, 2 * BB * DD);

  // xs = emb @ W_e2s^T + b      (bf16 out, TDM-staged A)
  gemm16k<true><<<dim3((BB * SS) / 16, 2), dim3(256), 0, stream>>>(
      embBF, wE2S, be2s, (void*)xsBF);

  // sequential recurrence (double-buffered TDM x_t prefetch)
  scan_kernel<<<dim3(1), dim3(1024), 0, stream>>>(
      xsBF, wT[0], wT[1], wT[2], wT[3], wT[4], wT[5], wT[6], wT[7], wT[8],
      lnw, lnb, gssb, lssb, outsBF);

  // out = outs @ W_s2o^T + b    (f32 out, TDM-staged A)
  gemm16k<false><<<dim3((BB * SS) / 16, 2), dim3(256), 0, stream>>>(
      outsBF, wS2O, bs2o, d_out);
}
